// SwitchHeadAttention_61667140436621
// MI455X (gfx1250) — matrix operand order repaired and checked by reading, not compiled, not run on hardware
//
#include <hip/hip_runtime.h>
#include <hip/hip_bf16.h>

// ---------------------------------------------------------------------------
// SwitchHead attention for MI455X (gfx1250, wave32, WMMA 16x16x32 f16).
// Pipeline:
//   1) cast kernels: x -> f16, [Wq|Wkv] -> f16 transposed (K-contiguous), Wo -> f16 transposed
//   2) qkv_gemm:     xh[4096,1024] @ WhT -> Q (scaled), K, V^T
//                    B tiles staged via GLOBAL_LOAD_ASYNC_TO_LDS (double-buffered)
//   3) flash_attn:   online-softmax attention, 8 wmma per 32-key step
//   4) out_proj:     per-head 64x64 projection + bias -> fp32 output
// ---------------------------------------------------------------------------

typedef __attribute__((ext_vector_type(16))) _Float16 v16h;
typedef __attribute__((ext_vector_type(8)))  _Float16 v8h;
typedef __attribute__((ext_vector_type(8)))  float    v8f;

#define DIM  1024
#define H    16
#define DH   64
#define BB   2
#define TT   2048
#define NQKV 3072
#define MROWS (BB * TT)        // 4096
#define SCALE 0.125f           // DH^-0.5

// ---- WMMA fragment helpers (layouts per CDNA5 ISA 7.12.2, wave32) ----------

// A-matrix 16x32 f16: lane l -> row (l&15); lanes 0-15 hold K=0..7,16..23,
// lanes 16-31 hold K=8..15,24..31. Two contiguous 16-byte loads per lane.
__device__ __forceinline__ v16h load_frag_a(const _Float16* __restrict__ p,
                                            int ld, int lane) {
  const int r  = lane & 15;
  const int kb = (lane >> 4) << 3;
  const _Float16* b0 = p + r * ld + kb;
  union { v16h v; v8h h[2]; } u;
  u.h[0] = *(const v8h*)(b0);
  u.h[1] = *(const v8h*)(b0 + 16);
  return u.v;
}

// B-matrix 32x16 f16 from a K-contiguous layout (column n has its K values
// contiguous, columns strided by ldk): lane l -> column (l&15);
// lanes 0-15 hold K=0..7,16..23, lanes 16-31 hold K=8..15,24..31.
__device__ __forceinline__ v16h load_frag_b(const _Float16* __restrict__ p,
                                            int ldk, int lane) {
  const int c  = lane & 15;
  const int kb = (lane >> 4) << 3;
  const _Float16* b0 = p + c * ldk + kb;
  union { v16h v; v8h h[2]; } u;
  u.h[0] = *(const v8h*)(b0);
  u.h[1] = *(const v8h*)(b0 + 16);
  return u.v;
}

__device__ __forceinline__ v8f wmma_f16(v16h a, v16h b, v8f c) {
  return __builtin_amdgcn_wmma_f32_16x16x32_f16(
      /*neg_a=*/false, a, /*neg_b=*/false, b,
      /*c_mod=*/(short)0, c, /*reuse_a=*/false, /*reuse_b=*/false);
}

// CDNA5 async global->LDS DMA (ASYNCcnt-tracked), GV addressing mode.
__device__ __forceinline__ void async_copy_b128(unsigned lds_off,
                                                const _Float16* gp) {
  asm volatile("global_load_async_to_lds_b128 %0, %1, off"
               :: "v"(lds_off), "v"(gp)
               : "memory");
}
__device__ __forceinline__ void wait_async0() {
  asm volatile("s_wait_asynccnt 0" ::: "memory");
}

// ---- 1) cast kernels -------------------------------------------------------

__global__ __launch_bounds__(256) void cast_x_f16(const float* __restrict__ x,
                                                  _Float16* __restrict__ xh,
                                                  int n) {
  int i = blockIdx.x * 256 + threadIdx.x;
  if (i < n) xh[i] = (_Float16)x[i];
}

// WhT[n][k] = W[k][n], n in [0,3072): n<1024 -> Wq, else Wkv (kv-major cols).
__global__ __launch_bounds__(256) void build_whT(const float* __restrict__ Wq,
                                                 const float* __restrict__ Wkv,
                                                 _Float16* __restrict__ whT) {
  int i = blockIdx.x * 256 + threadIdx.x;   // i < 3072*1024
  int n = i >> 10;
  int k = i & 1023;
  float v = (n < 1024) ? Wq[k * 1024 + n] : Wkv[k * 2048 + (n - 1024)];
  whT[i] = (_Float16)v;
}

// WoT[h][e][d] = Wo[h][d][e]
__global__ __launch_bounds__(256) void build_woT(const float* __restrict__ Wo,
                                                 _Float16* __restrict__ woT) {
  int i = blockIdx.x * 256 + threadIdx.x;   // i < 16*64*64
  int h = i >> 12;
  int e = (i >> 6) & 63;
  int d = i & 63;
  woT[i] = (_Float16)Wo[(h * 64 + d) * 64 + e];
}

// ---- 2) fused QKV GEMM -----------------------------------------------------
// grid: (NQKV/64, MROWS/128), block 256 (8 waves). Wave w -> 16-row strip.
// 64-col x 64-k B tiles staged into LDS with async DMA, double-buffered:
// next tile's DMA overlaps the current tile's 8 WMMAs.
__global__ __launch_bounds__(256) void qkv_gemm(
    const _Float16* __restrict__ xh, const _Float16* __restrict__ whT,
    _Float16* __restrict__ q, _Float16* __restrict__ kmat,
    _Float16* __restrict__ vT) {
  __shared__ _Float16 lds_b[2][64 * 64];     // 16 KB
  const int lane = threadIdx.x & 31;
  const int wave = threadIdx.x >> 5;
  const int n0   = blockIdx.x * 64;
  const int row0 = (blockIdx.y * 8 + wave) * 16;

  // stage one 64x64 tile: 8 KB, 256 threads x 2 x 16B async transfers
  auto stage = [&](int buf, int kk) {
#pragma unroll
    for (int j = 0; j < 2; ++j) {
      int chunk = threadIdx.x * 2 + j;       // 0..511
      int c  = chunk >> 3;                   // column 0..63
      int kp = (chunk & 7) << 3;             // k offset 0..56 step 8
      const _Float16* gp = &whT[(size_t)(n0 + c) * DIM + kk + kp];
      unsigned lds_off = (unsigned)(size_t)&lds_b[buf][c * 64 + kp];
      async_copy_b128(lds_off, gp);
    }
  };

  stage(0, 0);
  v8f acc[4] = {};
  for (int kk = 0; kk < DIM; kk += 64) {
    const int cur = (kk >> 6) & 1;
    wait_async0();                 // own DMA ops for current tile done
    __syncthreads();               // everyone's DMA landed; prev reads done
    if (kk + 64 < DIM) stage(cur ^ 1, kk + 64);   // overlap next DMA

    if (kk + 64 < DIM)
      __builtin_prefetch(xh + (size_t)row0 * DIM + kk + 64, 0, 3);
    const _Float16* bufp = &lds_b[cur][0];
    v16h a0 = load_frag_a(xh + (size_t)row0 * DIM + kk, DIM, lane);
    v16h a1 = load_frag_a(xh + (size_t)row0 * DIM + kk + 32, DIM, lane);
#pragma unroll
    for (int t = 0; t < 4; ++t) {
      v16h b0 = load_frag_b(bufp + (t * 16) * 64, 64, lane);
      v16h b1 = load_frag_b(bufp + (t * 16) * 64 + 32, 64, lane);
      acc[t] = wmma_f16(a0, b0, acc[t]);
      acc[t] = wmma_f16(a1, b1, acc[t]);
    }
  }

  // route output columns: [0,1024)->Q (pre-scaled), [1024,2048)->K, rest->V^T
  const int bidx = row0 >> 11;              // batch
  const int trow = row0 & (TT - 1);
  const int N  = lane & 15;
  const int hi = lane >> 4;
#pragma unroll
  for (int t = 0; t < 4; ++t) {
    int n = n0 + t * 16 + N;
#pragma unroll
    for (int r = 0; r < 8; ++r) {
      int M  = r + (hi << 3);
      int tw = trow + M;
      float val = acc[t][r];
      if (n < 1024) {
        int h = n >> 6, d = n & 63;
        q[(((size_t)(bidx * H + h)) * TT + tw) * DH + d] =
            (_Float16)(val * SCALE);
      } else if (n < 2048) {
        int n2 = n - 1024, h = n2 >> 6, d = n2 & 63;
        kmat[(((size_t)(bidx * H + h)) * TT + tw) * DH + d] = (_Float16)val;
      } else {
        int n2 = n - 2048, h = n2 >> 6, d = n2 & 63;
        vT[(((size_t)(bidx * H + h)) * DH + d) * TT + tw] = (_Float16)val;
      }
    }
  }
}

// ---- 3) flash attention ----------------------------------------------------
// One wave per (b,h, 16-query tile). Online softmax; row statistics live
// per-lane and match the 16x16 f32 C layout (VGPR r <-> row r / r+8).
__global__ __launch_bounds__(256) void flash_attn(
    const _Float16* __restrict__ q, const _Float16* __restrict__ kmat,
    const _Float16* __restrict__ vT, _Float16* __restrict__ attn) {
  __shared__ _Float16 lds_p[8][16 * 32];
  const int lane = threadIdx.x & 31;
  const int wave = threadIdx.x >> 5;
  _Float16* pbuf = lds_p[wave];

  const int wg = blockIdx.x * 8 + wave;
  const int bh = wg >> 7;                   // (b*H + h)
  const int qt = wg & 127;
  const _Float16* qp = q    + (size_t)bh * TT * DH + (size_t)qt * 16 * DH;
  const _Float16* kp = kmat + (size_t)bh * TT * DH;
  const _Float16* vp = vT   + (size_t)bh * DH * TT;

  const v16h aq0 = load_frag_a(qp, DH, lane);
  const v16h aq1 = load_frag_a(qp + 32, DH, lane);

  v8f o[4] = {};
  float mrun[8], lsum[8];
#pragma unroll
  for (int r = 0; r < 8; ++r) { mrun[r] = -3.0e38f; lsum[r] = 0.0f; }

  const int N  = lane & 15;
  const int hi = lane >> 4;

  for (int jb = 0; jb < TT; jb += 32) {
    // S = Q * K^T for keys [jb, jb+32): batch the 4 B loads, then 4 wmma
    v8f s0 = {}, s1 = {};
    {
      v16h b00 = load_frag_b(kp + (size_t)jb * DH, DH, lane);
      v16h b01 = load_frag_b(kp + (size_t)jb * DH + 32, DH, lane);
      v16h b10 = load_frag_b(kp + (size_t)(jb + 16) * DH, DH, lane);
      v16h b11 = load_frag_b(kp + (size_t)(jb + 16) * DH + 32, DH, lane);
      s0 = wmma_f16(aq0, b00, s0);
      s0 = wmma_f16(aq1, b01, s0);
      s1 = wmma_f16(aq0, b10, s1);
      s1 = wmma_f16(aq1, b11, s1);
    }
    // online softmax update
    float corr[8];
#pragma unroll
    for (int r = 0; r < 8; ++r) {
      float mx = fmaxf(s0[r], s1[r]);
#pragma unroll
      for (int msk = 8; msk >= 1; msk >>= 1)
        mx = fmaxf(mx, __shfl_xor(mx, msk, 32));
      float mnew = fmaxf(mrun[r], mx);
      corr[r] = __expf(mrun[r] - mnew);
      float p0 = __expf(s0[r] - mnew);
      float p1 = __expf(s1[r] - mnew);
      float rs = p0 + p1;
#pragma unroll
      for (int msk = 8; msk >= 1; msk >>= 1)
        rs += __shfl_xor(rs, msk, 32);
      lsum[r] = lsum[r] * corr[r] + rs;
      mrun[r] = mnew;
      s0[r] = p0;
      s1[r] = p1;
    }
#pragma unroll
    for (int t = 0; t < 4; ++t)
#pragma unroll
      for (int r = 0; r < 8; ++r) o[t][r] *= corr[r];

    // C-layout P -> LDS -> A-layout fragment
#pragma unroll
    for (int r = 0; r < 8; ++r) {
      int M = r + (hi << 3);
      pbuf[M * 32 + N]      = (_Float16)s0[r];
      pbuf[M * 32 + 16 + N] = (_Float16)s1[r];
    }
    v16h ap = load_frag_a(pbuf, 32, lane);

    // O += P * V  (V^T layout: columns d contiguous over t), 4 wmma
    v16h bv0 = load_frag_b(vp + (size_t)(0 * 16) * TT + jb, TT, lane);
    v16h bv1 = load_frag_b(vp + (size_t)(1 * 16) * TT + jb, TT, lane);
    v16h bv2 = load_frag_b(vp + (size_t)(2 * 16) * TT + jb, TT, lane);
    v16h bv3 = load_frag_b(vp + (size_t)(3 * 16) * TT + jb, TT, lane);
    o[0] = wmma_f16(ap, bv0, o[0]);
    o[1] = wmma_f16(ap, bv1, o[1]);
    o[2] = wmma_f16(ap, bv2, o[2]);
    o[3] = wmma_f16(ap, bv3, o[3]);
  }

  // normalize and store [bh][t][d] as f16 for the projection stage
  _Float16* op = attn + (size_t)bh * TT * DH + (size_t)qt * 16 * DH;
  float inv[8];
#pragma unroll
  for (int r = 0; r < 8; ++r) inv[r] = 1.0f / lsum[r];
#pragma unroll
  for (int t = 0; t < 4; ++t)
#pragma unroll
    for (int r = 0; r < 8; ++r) {
      int M = r + (hi << 3);
      op[(size_t)M * DH + t * 16 + N] = (_Float16)(o[t][r] * inv[r]);
    }
}

// ---- 4) per-head output projection + bias ---------------------------------
__global__ __launch_bounds__(256) void out_proj(
    const _Float16* __restrict__ attn, const _Float16* __restrict__ woT,
    const float* __restrict__ bo, float* __restrict__ out) {
  const int lane = threadIdx.x & 31;
  const int wave = threadIdx.x >> 5;
  const int wg = blockIdx.x * 8 + wave;
  const int bh = wg >> 7;
  const int t0 = (wg & 127) * 16;
  const int b  = bh >> 4;
  const int h  = bh & 15;

  const _Float16* ap = attn + (size_t)bh * TT * DH + (size_t)t0 * DH;
  const _Float16* wp = woT + (size_t)h * DH * DH;
  const v16h a0 = load_frag_a(ap, DH, lane);
  const v16h a1 = load_frag_a(ap + 32, DH, lane);

  v8f c[4] = {};
#pragma unroll
  for (int t = 0; t < 4; ++t) {
    v16h b0 = load_frag_b(wp + (size_t)(t * 16) * DH, DH, lane);
    v16h b1 = load_frag_b(wp + (size_t)(t * 16) * DH + 32, DH, lane);
    c[t] = wmma_f16(a0, b0, c[t]);
    c[t] = wmma_f16(a1, b1, c[t]);
  }

  const int N  = lane & 15;
  const int hi = lane >> 4;
#pragma unroll
  for (int t = 0; t < 4; ++t) {
    int e = t * 16 + N;
    float bias = bo[h * DH + e];
#pragma unroll
    for (int r = 0; r < 8; ++r) {
      int M = r + (hi << 3);
      out[((size_t)b * TT + t0 + M) * (H * DH) + h * DH + e] = c[t][r] + bias;
    }
  }
}

// ---- launch ----------------------------------------------------------------

extern "C" void kernel_launch(void* const* d_in, const int* in_sizes, int n_in,
                              void* d_out, int out_size, void* d_ws,
                              size_t ws_size, hipStream_t stream) {
  const float* x   = (const float*)d_in[0];
  const float* Wq  = (const float*)d_in[1];
  const float* Wkv = (const float*)d_in[2];
  const float* Wo  = (const float*)d_in[3];
  const float* bo  = (const float*)d_in[4];
  float* out = (float*)d_out;

  char* ws = (char*)d_ws;
  size_t off = 0;
  _Float16* xh  = (_Float16*)(ws + off); off += (size_t)MROWS * DIM * 2;   // 8 MB
  _Float16* whT = (_Float16*)(ws + off); off += (size_t)NQKV * DIM * 2;    // 6 MB
  _Float16* woT = (_Float16*)(ws + off); off += (size_t)H * DH * DH * 2;   // 128 KB
  _Float16* q    = (_Float16*)(ws + off); off += (size_t)BB * H * TT * DH * 2;
  _Float16* kmat = (_Float16*)(ws + off); off += (size_t)BB * H * TT * DH * 2;
  _Float16* vT   = (_Float16*)(ws + off); off += (size_t)BB * H * TT * DH * 2;
  _Float16* attn = (_Float16*)(ws + off); off += (size_t)BB * H * TT * DH * 2;

  // 1) casts / layout transforms
  cast_x_f16<<<(MROWS * DIM) / 256, 256, 0, stream>>>(x, xh, MROWS * DIM);
  build_whT<<<(NQKV * DIM) / 256, 256, 0, stream>>>(Wq, Wkv, whT);
  build_woT<<<(H * DH * DH) / 256, 256, 0, stream>>>(Wo, woT);

  // 2) fused QKV projection GEMM (async-staged, double-buffered)
  qkv_gemm<<<dim3(NQKV / 64, MROWS / 128), 256, 0, stream>>>(xh, whT, q, kmat, vT);

  // 3) flash attention (one wave per (b,h,16-query tile))
  flash_attn<<<(BB * H * (TT / 16)) / 8, 256, 0, stream>>>(q, kmat, vT, attn);

  // 4) per-head output projection + bias
  out_proj<<<(BB * H * (TT / 16)) / 8, 256, 0, stream>>>(attn, woT, bo, out);
}
